// NeuronAttention_18270790877580
// MI455X (gfx1250) — matrix operand order repaired and checked by reading, not compiled, not run on hardware
//
#include <hip/hip_runtime.h>

typedef unsigned short u16;
typedef __attribute__((ext_vector_type(16))) __bf16 bf16x16;
typedef __attribute__((ext_vector_type(8)))  float  floatx8;
typedef __attribute__((ext_vector_type(4)))  float  f32x4;
typedef __attribute__((ext_vector_type(8)))  u16    u16x8;
typedef __attribute__((ext_vector_type(4)))  u16    u16x4;
typedef int v4i __attribute__((vector_size(16)));

#define AS1 __attribute__((address_space(1)))
#define AS3 __attribute__((address_space(3)))

#if defined(__has_builtin)
#if __has_builtin(__builtin_amdgcn_global_load_async_to_lds_b128)
#define HAVE_ASYNC 1
#endif
#if __has_builtin(__builtin_amdgcn_s_wait_asynccnt)
#define HAVE_WAITA 1
#endif
#endif

union Frag {
    bf16x16 v;
    f32x4   f4[2];
};

__device__ __forceinline__ u16 f2bf(float f) {
    unsigned int u = __float_as_uint(f);
    unsigned int r = u + 0x7FFFu + ((u >> 16) & 1u);   // RNE
    return (u16)(r >> 16);
}

// 8-bf16 (16B) global -> LDS copy; async on CDNA5 when the builtin exists.
__device__ __forceinline__ void copy8_g2s(const u16* g, u16* s) {
#ifdef HAVE_ASYNC
    __builtin_amdgcn_global_load_async_to_lds_b128(
        (AS1 v4i*)(unsigned long long)g,
        (AS3 v4i*)(unsigned int)(unsigned long long)s, 0, 0);
#else
    *(u16x8*)s = *(const u16x8*)g;
#endif
}

__device__ __forceinline__ void wait_async_all() {
#ifdef HAVE_ASYNC
#ifdef HAVE_WAITA
    __builtin_amdgcn_s_wait_asynccnt(0);
#else
    asm volatile("s_wait_asynccnt 0x0" ::: "memory");
#endif
#endif
}

// ---------------------------------------------------------------------------
// Prep: elementwise fp32 -> bf16 (x)
// ---------------------------------------------------------------------------
__global__ __launch_bounds__(256) void convert_x(const float* __restrict__ x,
                                                 u16* __restrict__ xb) {
    long i = ((long)blockIdx.x * 256 + threadIdx.x) * 8;
    f32x4 a = *(const f32x4*)(x + i);
    f32x4 b = *(const f32x4*)(x + i + 4);
    u16x8 o = {f2bf(a.x), f2bf(a.y), f2bf(a.z), f2bf(a.w),
               f2bf(b.x), f2bf(b.y), f2bf(b.z), f2bf(b.w)};
    *(u16x8*)(xb + i) = o;
}

// ---------------------------------------------------------------------------
// Prep: W [K][N] fp32  ->  Wt [N][K] bf16   (32x32 LDS tile transpose)
// ---------------------------------------------------------------------------
__global__ __launch_bounds__(256) void transpose_convert(
    const float* __restrict__ W, u16* __restrict__ Wt, int K, int N) {
    __shared__ u16 Tl[32][33];
    int k0 = blockIdx.x * 32;
    int n0 = blockIdx.y * 32;
    int r  = threadIdx.x >> 3;
    int c4 = (threadIdx.x & 7) * 4;
    f32x4 v = *(const f32x4*)(W + (long)(k0 + r) * N + n0 + c4);
    Tl[r][c4 + 0] = f2bf(v.x); Tl[r][c4 + 1] = f2bf(v.y);
    Tl[r][c4 + 2] = f2bf(v.z); Tl[r][c4 + 3] = f2bf(v.w);
    __syncthreads();
    u16x4 o = {Tl[c4 + 0][r], Tl[c4 + 1][r], Tl[c4 + 2][r], Tl[c4 + 3][r]};
    *(u16x4*)(Wt + (long)(n0 + r) * K + k0 + c4) = o;
}

// ---------------------------------------------------------------------------
// Zero cache tail: slots [1024,2048) of updated_k / updated_v.
// ---------------------------------------------------------------------------
__global__ __launch_bounds__(256) void zero_tail(float* __restrict__ out) {
    int tid    = blockIdx.x * 256 + threadIdx.x;
    int within = tid & 32767;
    int region = tid >> 15;
    int t  = region >> 6;
    int bh = region & 63;
    float* base = out + (t ? 25165824 : 8388608) + ((long)bh * 2048 + 1024) * 128;
    f32x4 z = {0.f, 0.f, 0.f, 0.f};
    *(f32x4*)(base + (long)within * 4) = z;
}

// ---------------------------------------------------------------------------
// bf16 GEMM, fp32 accumulate: C = A[M][K] * Bt[N][K]^T + bias.
// Double-buffered async LDS staging. Block tile 128x128x32; 8 waves of 32x64.
// MODE 0: C row-major fp32.
// MODE 1: QKV epilogue -> q bf16 [B,H,S,128]; k/v fp32 caches [B,H,2048,128];
//         k bf16 [bh][s][d]; v bf16 transposed [bh][d][s].
// ---------------------------------------------------------------------------
template<int MODE>
__global__ __launch_bounds__(256) void gemm_bf16_wmma(
    const u16* __restrict__ A, const u16* __restrict__ Bt,
    const float* __restrict__ bias, float* __restrict__ C,
    u16* __restrict__ qbf, float* __restrict__ kout, float* __restrict__ vout,
    u16* __restrict__ kbf, u16* __restrict__ vbf, int M, int N, int K)
{
    __shared__ __align__(16) u16 As[2][128 * 32];
    __shared__ __align__(16) u16 Bs[2][128 * 32];

    const int t    = threadIdx.x;
    const int m0   = blockIdx.x * 128;
    const int n0   = blockIdx.y * 128;
    const int wave = t >> 5, lane = t & 31;
    const int half = lane >> 4, ln = lane & 15;
    const int wm   = (wave >> 1) * 32;
    const int wn   = (wave & 1)  * 64;

    floatx8 acc[2][4];
#pragma unroll
    for (int mt = 0; mt < 2; ++mt)
#pragma unroll
        for (int nt = 0; nt < 4; ++nt)
#pragma unroll
            for (int r = 0; r < 8; ++r) acc[mt][nt][r] = 0.f;

    const int rs = t >> 1;            // 0..127 (A row / Bt row)
    const int cs = (t & 1) * 16;      // 0 / 16

    auto stage = [&](int buf, int k0) {
        const u16* ga = A + (long)(m0 + rs) * K + k0 + cs;
        copy8_g2s(ga,     &As[buf][rs * 32 + cs]);
        copy8_g2s(ga + 8, &As[buf][rs * 32 + cs + 8]);
        const u16* gb = Bt + (long)(n0 + rs) * K + k0 + cs;
        copy8_g2s(gb,     &Bs[buf][rs * 32 + cs]);
        copy8_g2s(gb + 8, &Bs[buf][rs * 32 + cs + 8]);
    };

    stage(0, 0);
    int i = 0;
    for (int k0 = 0; k0 < K; k0 += 32, ++i) {
        wait_async_all();
        __syncthreads();
        if (k0 + 32 < K) stage((i + 1) & 1, k0 + 32);

        const u16* as = As[i & 1];
        const u16* bs = Bs[i & 1];
        Frag aF[2], bF[4];
#pragma unroll
        for (int mt = 0; mt < 2; ++mt) {
            int r = wm + mt * 16 + ln;
            aF[mt].f4[0] = *(const f32x4*)&as[r * 32 + half * 8];
            aF[mt].f4[1] = *(const f32x4*)&as[r * 32 + half * 8 + 16];
        }
#pragma unroll
        for (int nt = 0; nt < 4; ++nt) {
            int n = wn + nt * 16 + ln;
            bF[nt].f4[0] = *(const f32x4*)&bs[n * 32 + half * 8];
            bF[nt].f4[1] = *(const f32x4*)&bs[n * 32 + half * 8 + 16];
        }
#pragma unroll
        for (int mt = 0; mt < 2; ++mt)
#pragma unroll
            for (int nt = 0; nt < 4; ++nt)
                acc[mt][nt] = __builtin_amdgcn_wmma_f32_16x16x32_bf16(
                    false, aF[mt].v, false, bF[nt].v, (short)0, acc[mt][nt],
                    false, false);
    }

#pragma unroll
    for (int mt = 0; mt < 2; ++mt) {
#pragma unroll
        for (int nt = 0; nt < 4; ++nt) {
#pragma unroll
            for (int r = 0; r < 8; ++r) {
                int row = m0 + wm + mt * 16 + r + half * 8;
                int col = n0 + wn + nt * 16 + ln;
                float val = acc[mt][nt][r] + bias[col];
                if (MODE == 0) {
                    C[(long)row * N + col] = val;
                } else {
                    int part = col >> 12;
                    int wi   = col & 4095;
                    int h = wi >> 7, d = wi & 127;
                    int b = row >> 10, s = row & 1023;
                    int bh = b * 32 + h;
                    if (part == 0) {
                        qbf[((long)bh * 1024 + s) * 128 + d] = f2bf(val);
                    } else if (part == 1) {
                        kout[((long)bh * 2048 + s) * 128 + d] = val;
                        kbf[((long)bh * 1024 + s) * 128 + d]  = f2bf(val);
                    } else {
                        vout[((long)bh * 2048 + s) * 128 + d] = val;
                        vbf[((long)bh * 128 + d) * 1024 + s]  = f2bf(val);
                    }
                }
            }
        }
    }
}

// ---------------------------------------------------------------------------
// Flash attention: grid = B*H*8 q-blocks, 8 waves x 16 q rows. Double-buffered
// async staging of bf16 K [key][d] and pre-transposed V [d][key].
// ---------------------------------------------------------------------------
__global__ __launch_bounds__(256) void attn_kernel(
    const u16* __restrict__ qbf, const u16* __restrict__ kbf,
    const u16* __restrict__ vbf, u16* __restrict__ obf)
{
    __shared__ __align__(16) u16 Kl[2][32 * 128];
    __shared__ __align__(16) u16 Vl[2][128 * 32];
    __shared__ __align__(16) u16 Pl[8 * 16 * 32];

    const int t    = threadIdx.x;
    const int qb   = blockIdx.x & 7;
    const int bh   = blockIdx.x >> 3;
    const int w    = t >> 5, lane = t & 31;
    const int half = lane >> 4, ln = lane & 15;
    const int qr0  = qb * 128 + w * 16;

    Frag qfr[4];
    {
        long qbase = ((long)bh * 1024 + (qr0 + ln)) * 128;
#pragma unroll
        for (int ks = 0; ks < 4; ++ks) {
            int k0 = ks * 32 + half * 8;
            qfr[ks].f4[0] = *(const f32x4*)(qbf + qbase + k0);
            qfr[ks].f4[1] = *(const f32x4*)(qbf + qbase + k0 + 16);
        }
    }

    floatx8 O[8];
    float m_run[8], l_run[8];
#pragma unroll
    for (int dt = 0; dt < 8; ++dt)
#pragma unroll
        for (int r = 0; r < 8; ++r) O[dt][r] = 0.f;
#pragma unroll
    for (int r = 0; r < 8; ++r) { m_run[r] = -1e30f; l_run[r] = 0.f; }

    const int kend = qb * 128 + 128;
    const int skey = t >> 3, sck = (t & 7) * 16;   // K staging: [key][d]
    const int sd   = t >> 1, scv = (t & 1) * 16;   // V staging: [d][key]
    const float sc_qk = 0.08838834764831845f;      // 1/sqrt(128)

    auto stage = [&](int buf, int kc) {
        const u16* gk = kbf + ((long)bh * 1024 + kc + skey) * 128 + sck;
        copy8_g2s(gk,     &Kl[buf][skey * 128 + sck]);
        copy8_g2s(gk + 8, &Kl[buf][skey * 128 + sck + 8]);
        const u16* gv = vbf + ((long)bh * 128 + sd) * 1024 + kc + scv;
        copy8_g2s(gv,     &Vl[buf][sd * 32 + scv]);
        copy8_g2s(gv + 8, &Vl[buf][sd * 32 + scv + 8]);
    };

    stage(0, 0);
    int it = 0;
    for (int kc = 0; kc < kend; kc += 32, ++it) {
        wait_async_all();
        __syncthreads();
        if (kc + 32 < kend) stage((it + 1) & 1, kc + 32);
        const u16* kl = Kl[it & 1];
        const u16* vl = Vl[it & 1];

        floatx8 s[2];
#pragma unroll
        for (int nt = 0; nt < 2; ++nt) {
#pragma unroll
            for (int r = 0; r < 8; ++r) s[nt][r] = 0.f;
#pragma unroll
            for (int ks = 0; ks < 4; ++ks) {
                Frag bfr;
                int key = nt * 16 + ln;
                int kb  = ks * 32 + half * 8;
                bfr.f4[0] = *(const f32x4*)&kl[key * 128 + kb];
                bfr.f4[1] = *(const f32x4*)&kl[key * 128 + kb + 16];
                s[nt] = __builtin_amdgcn_wmma_f32_16x16x32_bf16(
                    false, qfr[ks].v, false, bfr.v, (short)0, s[nt], false, false);
            }
        }

#pragma unroll
        for (int r = 0; r < 8; ++r) {
            int qi = qr0 + r + half * 8;
            float s0 = s[0][r] * sc_qk; if (kc + ln      > qi) s0 = -1e30f;
            float s1 = s[1][r] * sc_qk; if (kc + 16 + ln > qi) s1 = -1e30f;
            float mx = fmaxf(s0, s1);
            mx = fmaxf(mx, __shfl_xor(mx, 1, 32));
            mx = fmaxf(mx, __shfl_xor(mx, 2, 32));
            mx = fmaxf(mx, __shfl_xor(mx, 4, 32));
            mx = fmaxf(mx, __shfl_xor(mx, 8, 32));
            float mnew  = fmaxf(m_run[r], mx);
            float scale = __expf(m_run[r] - mnew);
            m_run[r] = mnew;
            float p0 = __expf(s0 - mnew);
            float p1 = __expf(s1 - mnew);
            float rsum = p0 + p1;
            rsum += __shfl_xor(rsum, 1, 32);
            rsum += __shfl_xor(rsum, 2, 32);
            rsum += __shfl_xor(rsum, 4, 32);
            rsum += __shfl_xor(rsum, 8, 32);
            l_run[r] = l_run[r] * scale + rsum;
#pragma unroll
            for (int dt = 0; dt < 8; ++dt) O[dt][r] *= scale;
            int prow = (w * 16 + r + half * 8) * 32;
            Pl[prow + ln]      = f2bf(p0);
            Pl[prow + 16 + ln] = f2bf(p1);
        }

        Frag pa;
        pa.f4[0] = *(const f32x4*)&Pl[(w * 16 + ln) * 32 + half * 8];
        pa.f4[1] = *(const f32x4*)&Pl[(w * 16 + ln) * 32 + half * 8 + 16];

#pragma unroll
        for (int dt = 0; dt < 8; ++dt) {
            Frag bfr;
            int d = dt * 16 + ln;
            bfr.f4[0] = *(const f32x4*)&vl[d * 32 + half * 8];
            bfr.f4[1] = *(const f32x4*)&vl[d * 32 + half * 8 + 16];
            O[dt] = __builtin_amdgcn_wmma_f32_16x16x32_bf16(
                false, pa.v, false, bfr.v, (short)0, O[dt], false, false);
        }
    }

    const int b = bh >> 5, h = bh & 31;
#pragma unroll
    for (int dt = 0; dt < 8; ++dt) {
#pragma unroll
        for (int r = 0; r < 8; ++r) {
            int qi = qr0 + r + half * 8;
            int d  = dt * 16 + ln;
            obf[((long)(b * 1024 + qi)) * 4096 + h * 128 + d] =
                f2bf(O[dt][r] / l_run[r]);
        }
    }
}

// ---------------------------------------------------------------------------
extern "C" void kernel_launch(void* const* d_in, const int* in_sizes, int n_in,
                              void* d_out, int out_size, void* d_ws, size_t ws_size,
                              hipStream_t stream) {
    const float* x     = (const float*)d_in[0];
    const float* W_qkv = (const float*)d_in[2];
    const float* b_qkv = (const float*)d_in[3];
    const float* W_out = (const float*)d_in[4];
    const float* b_out = (const float*)d_in[5];

    float* y    = (float*)d_out;
    float* kout = y + 8388608;
    float* vout = y + 25165824;

    char* ws = (char*)d_ws;
    u16* q_bf = (u16*)(ws);                            // 16 MB  [bh][s][128]
    u16* k_bf = (u16*)(ws + 16777216);                 // 16 MB  [bh][s][128]
    u16* v_bf = (u16*)(ws + 33554432);                 // 16 MB  [bh][128][s]
    u16* o_bf = (u16*)(ws + 50331648);                 // 16 MB  [2048][4096]
    u16* x_bf = (u16*)(ws + 67108864);                 // 16 MB  [2048][4096]
    u16* Wqt  = (u16*)(ws + 83886080);                 // 96 MB  [12288][4096]
    u16* Wot  = (u16*)(ws + 184549376);                // 32 MB  [4096][4096]

    convert_x<<<dim3(4096), dim3(256), 0, stream>>>(x, x_bf);
    transpose_convert<<<dim3(128, 384), dim3(256), 0, stream>>>(W_qkv, Wqt, 4096, 12288);
    transpose_convert<<<dim3(128, 128), dim3(256), 0, stream>>>(W_out, Wot, 4096, 4096);
    zero_tail<<<dim3(16384), dim3(256), 0, stream>>>(y);

    gemm_bf16_wmma<1><<<dim3(16, 96), dim3(256), 0, stream>>>(
        x_bf, Wqt, b_qkv, (float*)nullptr, q_bf, kout, vout, k_bf, v_bf,
        2048, 12288, 4096);

    attn_kernel<<<dim3(512), dim3(256), 0, stream>>>(q_bf, k_bf, v_bf, o_bf);

    gemm_bf16_wmma<0><<<dim3(16, 32), dim3(256), 0, stream>>>(
        o_bf, Wot, b_out, y, (u16*)nullptr, (float*)nullptr, (float*)nullptr,
        (u16*)nullptr, (u16*)nullptr, 2048, 4096, 4096);
}
